// RecurrentNonLocalKmeansAttentionWithLoss_6133213299491
// MI455X (gfx1250) — matrix operand order repaired and checked by reading, not compiled, not run on hardware
//
#include <hip/hip_runtime.h>

// ---------------------------------------------------------------------------
// Problem constants (reference: N=8, CH=256, H=W=96, C=64, NCL=128, WIN=144)
// ---------------------------------------------------------------------------
#define N_BATCH 8
#define CH_ 256
#define H_ 96
#define W_ 96
#define L_ 9216          // H*W
#define C_ 64            // qk dim
#define NCL_ 128
#define WIN_ 144
#define NB_ 64           // L / WIN
#define KEYS_ 432        // 3*WIN
#define RES_ 0.1f

typedef __attribute__((ext_vector_type(16))) __bf16 bf16x16;
typedef __attribute__((ext_vector_type(8)))  float  f32x8;

union B8 { __bf16 h[8]; uint4 v; };   // 16B-aligned 8x bf16 shuttle

__device__ __forceinline__ unsigned pack2(__bf16 a, __bf16 b) {
  union { __bf16 h[2]; unsigned u; } p;
  p.h[0] = a; p.h[1] = b;
  return p.u;
}

// ---------------------------------------------------------------------------
// WMMA fragment helpers (CDNA5 16x16x32 bf16, wave32; A/B in contiguous k order, C/D layout per ISA 7.12.2)
// A 16x32 (MxK), LDS tile [m][k]: lane m=lane&15, half=lane>>4;
//   elem e <-> k=16*half+e  (one 32B contiguous run)
// B 32x16 (KxN), LDS tile stored K-contiguous as [n][k]: elem e <-> k=16*half+e
//   => one 32B contiguous run per lane (2x ds_load_b128)
// C/D 16x16 f32: elem r <-> row m=r+8*half, col n=lane&15.
// ---------------------------------------------------------------------------
__device__ __forceinline__ bf16x16 load_a_lds(const __bf16* base, int row0, int ld, int k0) {
  const int lane = threadIdx.x & 31;
  const int m = lane & 15, half = lane >> 4;
  const __bf16* r = base + (size_t)(row0 + m) * ld + k0;
  bf16x16 a;
#pragma unroll
  for (int e = 0; e < 8; ++e) a[e] = r[16 * half + e];
#pragma unroll
  for (int e = 0; e < 8; ++e) a[8 + e] = r[16 * half + 8 + e];
  return a;
}

__device__ __forceinline__ bf16x16 load_b_rowK(const __bf16* base, int n0, int ld, int k0) {
  const int lane = threadIdx.x & 31;
  const __bf16* p = base + (size_t)(n0 + (lane & 15)) * ld + k0 + 16 * (lane >> 4);
  bf16x16 b;
#pragma unroll
  for (int e = 0; e < 16; ++e) b[e] = p[e];
  return b;
}

__device__ __forceinline__ f32x8 wmma_bf16(bf16x16 a, bf16x16 b, f32x8 c) {
  return __builtin_amdgcn_wmma_f32_16x16x32_bf16(false, a, false, b, (short)0, c, false, false);
}

__device__ __forceinline__ f32x8 zero8() {
  f32x8 z;
#pragma unroll
  for (int r = 0; r < 8; ++r) z[r] = 0.0f;
  return z;
}

// ---------------------------------------------------------------------------
// Kernel 1: conv3x3 (pad 1), 256 -> 64 channels, output x_embed (N,L,64) bf16
// One block per (n, image row h). 6 waves, each owns 16 pixels x 64 co.
// K-loop: (kh, ci-chunk of 32), inner kw 0..2 via shifted LDS A reads.
// ---------------------------------------------------------------------------
__global__ __launch_bounds__(192) void k_conv3x3(const float* __restrict__ x,
                                                 const float* __restrict__ w,
                                                 const float* __restrict__ bias,
                                                 __bf16* __restrict__ xe) {
  __shared__ __bf16 sX[98 * 32];        // padded row: [pixel+1][ci_local]
  __shared__ __bf16 sW[3 * 64 * 32];    // [kw][co][ci_local]  (B: [n][k])
  const int n = blockIdx.y, h = blockIdx.x;
  const int tid = threadIdx.x;
  const int wv = tid >> 5, lane = tid & 31, col = lane & 15, half = lane >> 4;
  const int m0 = wv * 16;

  f32x8 acc[4];
#pragma unroll
  for (int t = 0; t < 4; ++t) acc[t] = zero8();

  for (int kh = 0; kh < 3; ++kh) {
    const int hin = h + kh - 1;
    const bool hok = (hin >= 0) && (hin < H_);
    for (int cc = 0; cc < 8; ++cc) {
      // stage padded input row: pack 2 channels -> b32 LDS stores, global coalesced
      for (int i = tid; i < 98 * 16; i += 192) {
        const int kp = i / 98, pp = i % 98;
        const int win = pp - 1;
        __bf16 va = (__bf16)0.0f, vb = (__bf16)0.0f;
        if (hok && win >= 0 && win < W_) {
          const size_t base = (((size_t)n * CH_ + cc * 32 + 2 * kp) * H_ + hin) * W_ + win;
          va = (__bf16)x[base];
          vb = (__bf16)x[base + (size_t)H_ * W_];
        }
        *(unsigned*)&sX[pp * 32 + 2 * kp] = pack2(va, vb);
      }
      // stage weights K-contiguous: per-thread 8-element run -> b128 store
      for (int i = tid; i < 768; i += 192) {     // 3 kw * 64 co * 4 segs
        const int seg = i & 3;
        const int rest = i >> 2;
        const int co = rest & 63, kw = rest >> 6;
        B8 tmp;
#pragma unroll
        for (int q = 0; q < 8; ++q)
          tmp.h[q] = (__bf16)w[(((size_t)co * CH_ + cc * 32 + seg * 8 + q) * 3 + kh) * 3 + kw];
        *(uint4*)&sW[(kw * 64 + co) * 32 + seg * 8] = tmp.v;
      }
      __syncthreads();
#pragma unroll
      for (int kw = 0; kw < 3; ++kw) {
        bf16x16 a = load_a_lds(sX, m0 + kw, 32, 0);
#pragma unroll
        for (int t = 0; t < 4; ++t) {
          bf16x16 b = load_b_rowK(sW + kw * 64 * 32, t * 16, 32, 0);
          acc[t] = wmma_bf16(a, b, acc[t]);
        }
      }
      __syncthreads();
    }
  }
#pragma unroll
  for (int t = 0; t < 4; ++t) {
    const int co = t * 16 + col;
    const float bv = bias[co];
#pragma unroll
    for (int r = 0; r < 8; ++r) {
      const int p = m0 + r + 8 * half;  // pixel within row, 0..95
      xe[((size_t)n * L_ + h * W_ + p) * C_ + co] = (__bf16)(acc[t][r] + bv);
    }
  }
}

// ---------------------------------------------------------------------------
// Kernel 2: conv1x1 (256->256) GEMM, output y_embed (N,L,256) bf16
// Block: 64-pixel tile, 8 waves (4 M-tiles x 2 co-halves of 128).
// ---------------------------------------------------------------------------
__global__ __launch_bounds__(256) void k_conv1x1(const float* __restrict__ x,
                                                 const float* __restrict__ w,
                                                 const float* __restrict__ bias,
                                                 __bf16* __restrict__ ye) {
  __shared__ __bf16 sX[64 * 32];     // [pixel][ci_local]   (A: [m][k])
  __shared__ __bf16 sW[256 * 32];    // [co][ci_local]      (B: [n][k])
  const int n = blockIdx.y;
  const int l0 = blockIdx.x * 64;
  const int tid = threadIdx.x;
  const int wv = tid >> 5, lane = tid & 31, col = lane & 15, half = lane >> 4;
  const int m0 = (wv & 3) * 16;
  const int ct = wv >> 2;  // co half: 0 or 1 (base ct*128)

  f32x8 acc[8];
#pragma unroll
  for (int t = 0; t < 8; ++t) acc[t] = zero8();

  for (int cc = 0; cc < 8; ++cc) {
    // input tile: pack 2 channels -> b32 stores, reads coalesced over pixels
    for (int i = tid; i < 1024; i += 256) {
      const int kp = i >> 6, p = i & 63;
      const size_t base = ((size_t)n * CH_ + cc * 32 + 2 * kp) * (size_t)L_ + l0 + p;
      const float va = x[base];
      const float vb = x[base + (size_t)L_];
      *(unsigned*)&sX[p * 32 + 2 * kp] = pack2((__bf16)va, (__bf16)vb);
      if (cc < 7)  // CDNA5 global_prefetch_b8 for next K-chunk
        __builtin_prefetch(&x[base + 32 * (size_t)L_], 0, 1);
    }
    // weights K-contiguous: 8-element runs -> b128 stores
    for (int i = tid; i < 1024; i += 256) {
      const int seg = i & 3, co = i >> 2;
      B8 tmp;
#pragma unroll
      for (int q = 0; q < 8; ++q)
        tmp.h[q] = (__bf16)w[(size_t)co * CH_ + cc * 32 + seg * 8 + q];
      *(uint4*)&sW[co * 32 + seg * 8] = tmp.v;
    }
    __syncthreads();
    bf16x16 a = load_a_lds(sX, m0, 32, 0);
#pragma unroll
    for (int t = 0; t < 8; ++t) {
      bf16x16 b = load_b_rowK(sW, ct * 128 + t * 16, 32, 0);
      acc[t] = wmma_bf16(a, b, acc[t]);
    }
    __syncthreads();
  }
#pragma unroll
  for (int t = 0; t < 8; ++t) {
    const int co = ct * 128 + t * 16 + col;
    const float bv = bias[co];
#pragma unroll
    for (int r = 0; r < 8; ++r) {
      const int p = m0 + r + 8 * half;
      ye[((size_t)n * L_ + l0 + p) * CH_ + co] = (__bf16)(acc[t][r] + bv);
    }
  }
}

// ---------------------------------------------------------------------------
// Kernel 3: bucket assignment + row norms.
// argmax over cluster dots is scale-invariant => skip the 1e-12 normalize.
// ---------------------------------------------------------------------------
__global__ __launch_bounds__(256) void k_assign(const __bf16* __restrict__ xe,
                                                const float* __restrict__ means,
                                                int* __restrict__ bucket,
                                                float* __restrict__ normv) {
  const int g = blockIdx.x * 256 + threadIdx.x;
  if (g >= N_BATCH * L_) return;
  const __bf16* row = xe + (size_t)g * C_;
  float xr[C_];
  float ss = 0.0f;
#pragma unroll
  for (int k = 0; k < C_; ++k) {
    xr[k] = (float)row[k];
    ss += xr[k] * xr[k];
  }
  normv[g] = sqrtf(ss);
  int best = 0;
  float bd = -3.0e38f;
  for (int c = 0; c < NCL_; ++c) {
    float d = 0.0f;
#pragma unroll
    for (int k = 0; k < C_; ++k) d += xr[k] * means[(size_t)c * C_ + k];
    if (d > bd) { bd = d; best = c; }
  }
  bucket[g] = best;
}

// ---------------------------------------------------------------------------
// Kernels 4-6: deterministic STABLE counting sort of bucket codes.
// counts layout: [n][chunk=256][bin=128], chunk covers 36 consecutive pixels.
// ---------------------------------------------------------------------------
__global__ __launch_bounds__(256) void k_count(const int* __restrict__ bucket,
                                               int* __restrict__ counts) {
  const int n = blockIdx.x, ch = threadIdx.x;
  int* row = counts + ((size_t)n * 256 + ch) * NCL_;
  for (int b = 0; b < NCL_; ++b) row[b] = 0;
  const int* bk = bucket + (size_t)n * L_ + ch * 36;
  for (int i = 0; i < 36; ++i) row[bk[i]] += 1;
}

__global__ __launch_bounds__(128) void k_scan(int* __restrict__ counts) {
  __shared__ int tot[NCL_];
  __shared__ int off[NCL_];
  const int n = blockIdx.x, b = threadIdx.x;  // 128 threads = bins
  int* base = counts + (size_t)n * 256 * NCL_;
  int run = 0;
  for (int c = 0; c < 256; ++c) {
    const int t = base[c * NCL_ + b];
    base[c * NCL_ + b] = run;
    run += t;
  }
  tot[b] = run;
  __syncthreads();
  if (b == 0) {
    int r = 0;
    for (int i = 0; i < NCL_; ++i) { off[i] = r; r += tot[i]; }
  }
  __syncthreads();
  const int o = off[b];
  for (int c = 0; c < 256; ++c) base[c * NCL_ + b] += o;
}

__global__ __launch_bounds__(256) void k_scatter(const int* __restrict__ bucket,
                                                 int* __restrict__ counts,
                                                 int* __restrict__ indices) {
  const int n = blockIdx.x, ch = threadIdx.x;
  int* row = counts + ((size_t)n * 256 + ch) * NCL_;
  const int* bk = bucket + (size_t)n * L_;
  for (int i = 0; i < 36; ++i) {
    const int l = ch * 36 + i;
    const int b = bk[l];
    const int pos = row[b];
    row[b] = pos + 1;
    indices[(size_t)n * L_ + pos] = l;
  }
}

// ---------------------------------------------------------------------------
// Kernel 7: bucketed flash attention.
// Block = (bucket b, batch n), 9 waves x 16 query rows = 144 queries.
// 432 keys (self + prev + next window, rolled over 64 buckets) streamed in
// 32-key tiles; keys normalized by 1/max(||x||, 5e-5); V = raw y_embed.
// K tile stored [key][ci] (natural gather), V tile stored transposed [co][key]
// via packed b32 stores so every WMMA B fragment is 2x ds_load_b128.
// Output scattered to original pixel order (undo_sort) into ret (N,L,256) f32.
// ---------------------------------------------------------------------------
__global__ __launch_bounds__(288) void k_attn(const __bf16* __restrict__ xe,
                                              const __bf16* __restrict__ ye,
                                              const float* __restrict__ normv,
                                              const int* __restrict__ indices,
                                              float* __restrict__ ret) {
  __shared__ __bf16 sKt[32 * C_];        // [key][ci]   (B for QK^T: [n][k])
  __shared__ __bf16 sVt[CH_ * 32];       // [co][key]   (B for PV:   [n][k])
  __shared__ __bf16 sP[9 * 16 * 32];     // per-wave P tile (A: [m][k])
  __shared__ int    sL[32];
  __shared__ float  sS[32];

  const int n = blockIdx.y, b = blockIdx.x;
  const int tid = threadIdx.x;
  const int wv = tid >> 5, lane = tid & 31, col = lane & 15, half = lane >> 4;
  const size_t nL = (size_t)n * L_;

  // ---- query fragments (gathered directly from global, per-lane rows) ----
  const int qpos = b * WIN_ + wv * 16 + col;
  const int ql = indices[nL + qpos];
  const __bf16* qrow = xe + (nL + ql) * C_;
  bf16x16 aQ0, aQ1;
#pragma unroll
  for (int e = 0; e < 8; ++e) {
    aQ0[e]     = qrow[16 * half + e];
    aQ0[8 + e] = qrow[8 + 16 * half + e];
    aQ1[e]     = qrow[32 + 16 * half + e];
    aQ1[8 + e] = qrow[40 + 16 * half + e];
  }

  f32x8 o[16];
#pragma unroll
  for (int t = 0; t < 16; ++t) o[t] = zero8();
  float rmax[8], rsum[8];
#pragma unroll
  for (int r = 0; r < 8; ++r) { rmax[r] = -1.0e30f; rsum[r] = 0.0f; }

  __bf16* sPw = sP + wv * (16 * 32);
  unsigned* sVt32 = (unsigned*)sVt;

  for (int j0 = 0; j0 < KEYS_; j0 += 32) {
    // ---- key metadata: rolled-window gather + normalize scale ----
    if (tid < 32) {
      const int j = j0 + tid;
      int l = 0;
      float sc = 0.0f;
      if (j < KEYS_) {
        const int kb = (j < WIN_) ? b : (j < 2 * WIN_) ? ((b + NB_ - 1) & (NB_ - 1))
                                                       : ((b + 1) & (NB_ - 1));
        const int r = (j < WIN_) ? j : (j < 2 * WIN_) ? (j - WIN_) : (j - 2 * WIN_);
        l = indices[nL + kb * WIN_ + r];
        sc = 1.0f / fmaxf(normv[nL + l], 5e-5f);
      }
      sL[tid] = l;
      sS[tid] = sc;
    }
    __syncthreads();
    // ---- K tile [key][ci]: normalized row copies, b128 stores ----
    if (tid < 256) {
      const int key = tid >> 3, seg = tid & 7;
      const bool ok = (j0 + key) < KEYS_;
      const __bf16* src = xe + (nL + sL[key]) * C_ + seg * 8;
      const float sc = ok ? sS[key] : 0.0f;
      B8 tmp;
#pragma unroll
      for (int q = 0; q < 8; ++q)
        tmp.h[q] = ok ? (__bf16)((float)src[q] * sc) : (__bf16)0.0f;
      *(uint4*)&sKt[key * C_ + seg * 8] = tmp.v;
    }
    // ---- V tile transposed [co][key]: key-pair pack -> b32 stores ----
    for (int i = tid; i < 512; i += 288) {   // 16 key-pairs x 32 co-segments
      const int kp = i >> 5;                 // key pair (even-aligned validity)
      const int seg = i & 31;                // 8-channel segment
      uint4 va = make_uint4(0u, 0u, 0u, 0u), vb = va;
      if ((j0 + 2 * kp) < KEYS_) {
        va = ((const uint4*)(ye + (nL + sL[2 * kp]) * CH_))[seg];
        vb = ((const uint4*)(ye + (nL + sL[2 * kp + 1]) * CH_))[seg];
      }
      const __bf16* pa = (const __bf16*)&va;
      const __bf16* pb = (const __bf16*)&vb;
#pragma unroll
      for (int q = 0; q < 8; ++q) {
        const int co = seg * 8 + q;
        sVt32[co * 16 + kp] = pack2(pa[q], pb[q]);
      }
    }
    __syncthreads();

    // ---- S = Q x K^T : two 16x16 logits fragments, K-dim 64 = 2 wmma each --
    f32x8 s0 = zero8(), s1 = zero8();
    {
      bf16x16 bk0 = load_b_rowK(sKt, 0, C_, 0);
      s0 = wmma_bf16(aQ0, bk0, s0);
      bf16x16 bk1 = load_b_rowK(sKt, 0, C_, 32);
      s0 = wmma_bf16(aQ1, bk1, s0);
      bf16x16 bk2 = load_b_rowK(sKt, 16, C_, 0);
      s1 = wmma_bf16(aQ0, bk2, s1);
      bf16x16 bk3 = load_b_rowK(sKt, 16, C_, 32);
      s1 = wmma_bf16(aQ1, bk3, s1);
    }

    // ---- online softmax (row stats reduced across the 16 lanes per half) --
    const bool ok0 = (j0 + col) < KEYS_;
    const bool ok1 = (j0 + 16 + col) < KEYS_;
    float corr[8];
#pragma unroll
    for (int r = 0; r < 8; ++r) {
      float v0 = ok0 ? s0[r] : -1.0e30f;
      float v1 = ok1 ? s1[r] : -1.0e30f;
      float v = fmaxf(v0, v1);
      v = fmaxf(v, __shfl_xor(v, 1));
      v = fmaxf(v, __shfl_xor(v, 2));
      v = fmaxf(v, __shfl_xor(v, 4));
      v = fmaxf(v, __shfl_xor(v, 8));
      const float mnew = fmaxf(rmax[r], v);
      corr[r] = __expf(rmax[r] - mnew);
      const float p0 = ok0 ? __expf(s0[r] - mnew) : 0.0f;
      const float p1 = ok1 ? __expf(s1[r] - mnew) : 0.0f;
      float ps = p0 + p1;
      ps += __shfl_xor(ps, 1);
      ps += __shfl_xor(ps, 2);
      ps += __shfl_xor(ps, 4);
      ps += __shfl_xor(ps, 8);
      rsum[r] = rsum[r] * corr[r] + ps;
      rmax[r] = mnew;
      const int m = r + 8 * half;
      sPw[m * 32 + col]      = (__bf16)p0;
      sPw[m * 32 + 16 + col] = (__bf16)p1;
    }
#pragma unroll
    for (int t = 0; t < 16; ++t)
#pragma unroll
      for (int r = 0; r < 8; ++r) o[t][r] *= corr[r];

    asm volatile("s_wait_dscnt 0" ::: "memory");  // P tile visible to own wave

    // ---- O += P x V over the full 256 output channels (16 wmma) ----
    bf16x16 aP = load_a_lds(sPw, 0, 32, 0);
#pragma unroll
    for (int t = 0; t < 16; ++t) {
      bf16x16 bv = load_b_rowK(sVt, t * 16, 32, 0);
      o[t] = wmma_bf16(aP, bv, o[t]);
    }
    __syncthreads();  // protect sKt/sVt/sL before next staging round
  }

  // ---- normalize + scatter to original pixel order (undo_sort) ----
  float inv[8];
  int lrow[8];
#pragma unroll
  for (int r = 0; r < 8; ++r) {
    inv[r] = 1.0f / rsum[r];
    lrow[r] = indices[nL + b * WIN_ + wv * 16 + r + 8 * half];
  }
#pragma unroll
  for (int t = 0; t < 16; ++t) {
    const int co = t * 16 + col;
#pragma unroll
    for (int r = 0; r < 8; ++r)
      ret[(nL + lrow[r]) * CH_ + co] = o[t][r] * inv[r];
  }
}

// ---------------------------------------------------------------------------
// Kernel 8: (N,L,CH) -> NCHW transpose + residual: out = ret*0.1 + input_x
// ---------------------------------------------------------------------------
__global__ __launch_bounds__(256) void k_epilogue(const float* __restrict__ ret,
                                                  const float* __restrict__ x,
                                                  float* __restrict__ out) {
  __shared__ float tile[32][33];
  const int n = blockIdx.z;
  const int c0 = blockIdx.y * 32;
  const int l0 = blockIdx.x * 32;
  const int tx = threadIdx.x & 31, ty = threadIdx.x >> 5;
  for (int i = ty; i < 32; i += 8)
    tile[i][tx] = ret[((size_t)n * L_ + l0 + i) * CH_ + c0 + tx];
  __syncthreads();
  for (int i = ty; i < 32; i += 8) {
    const size_t idx = ((size_t)n * CH_ + c0 + i) * (size_t)L_ + l0 + tx;
    out[idx] = tile[tx][i] * RES_ + x[idx];
  }
}

// ---------------------------------------------------------------------------
// Host launcher
// ---------------------------------------------------------------------------
extern "C" void kernel_launch(void* const* d_in, const int* in_sizes, int n_in,
                              void* d_out, int out_size, void* d_ws, size_t ws_size,
                              hipStream_t stream) {
  const float* x     = (const float*)d_in[0];  // (8,256,96,96)
  const float* means = (const float*)d_in[1];  // (1,128,64)
  const float* wm    = (const float*)d_in[2];  // (64,256,3,3)
  const float* bm    = (const float*)d_in[3];  // (64,)
  const float* wa    = (const float*)d_in[4];  // (256,256,1,1)
  const float* ba    = (const float*)d_in[5];  // (256,)
  float* out = (float*)d_out;

  char* p = (char*)d_ws;
  auto carve = [&](size_t bytes) {
    char* r = p;
    p += (bytes + 255) & ~(size_t)255;
    return r;
  };
  __bf16* xe    = (__bf16*)carve((size_t)N_BATCH * L_ * C_ * 2);     // 9.4 MB
  __bf16* ye    = (__bf16*)carve((size_t)N_BATCH * L_ * CH_ * 2);    // 37.7 MB
  float*  normv = (float*)carve((size_t)N_BATCH * L_ * 4);
  int*    bkt   = (int*)carve((size_t)N_BATCH * L_ * 4);
  int*    idx   = (int*)carve((size_t)N_BATCH * L_ * 4);
  int*    cnts  = (int*)carve((size_t)N_BATCH * 256 * NCL_ * 4);     // 1 MB
  float*  ret   = (float*)carve((size_t)N_BATCH * L_ * CH_ * 4);     // 75.5 MB

  k_conv3x3<<<dim3(H_, N_BATCH), 192, 0, stream>>>(x, wm, bm, xe);
  k_conv1x1<<<dim3(L_ / 64, N_BATCH), 256, 0, stream>>>(x, wa, ba, ye);
  k_assign<<<dim3((N_BATCH * L_) / 256), 256, 0, stream>>>(xe, means, bkt, normv);
  k_count<<<dim3(N_BATCH), 256, 0, stream>>>(bkt, cnts);
  k_scan<<<dim3(N_BATCH), 128, 0, stream>>>(cnts);
  k_scatter<<<dim3(N_BATCH), 256, 0, stream>>>(bkt, cnts, idx);
  k_attn<<<dim3(NB_, N_BATCH), 288, 0, stream>>>(xe, ye, normv, idx, ret);
  k_epilogue<<<dim3(L_ / 32, CH_ / 32, N_BATCH), 256, 0, stream>>>(ret, x, out);
}
